// TestModel_58892591563210
// MI455X (gfx1250) — compile-verified, matrix operand order checked by reading, not run on hardware
//
#include <hip/hip_runtime.h>
#include <hip/hip_bf16.h>

// ---------------------------------------------------------------------------
// 10x [ linear(1024->4096)+LN ; linear(4096->1024)+LN ] on x[8,2048,1024]
// M = 16384. GEMMs via v_wmma_f32_16x16x32_bf16 (bf16 operands, f32 acc).
// Tile staging via async global->LDS DMA (ASYNCcnt) with double buffering.
// ---------------------------------------------------------------------------

typedef __attribute__((ext_vector_type(16))) __bf16 v16bf;
typedef __attribute__((ext_vector_type(8)))  __bf16 v8bf;
typedef __attribute__((ext_vector_type(8)))  float  v8f;

#define TM 128
#define TN 128
#define TK 32
#define LSTR 40   // LDS row stride in bf16 elements (32 + 8 pad, keeps 16B align)

// LDS byte-offset (32-bit) of a generic pointer into shared memory.
static __device__ __forceinline__ uint32_t lds_off32(const void* p) {
    return (uint32_t)(uintptr_t)(__attribute__((address_space(3))) const void*)p;
}

// 16B async DMA: global -> LDS, tracked by ASYNCcnt.
static __device__ __forceinline__ void async_ld_b128(const void* g, uint32_t lds) {
    asm volatile("global_load_async_to_lds_b128 %0, %1, off"
                 :: "v"(lds), "v"(g) : "memory");
}

// --------------------------- fp32 -> bf16 cast -----------------------------
__global__ __launch_bounds__(256) void cvt_f32_bf16(const float* __restrict__ in,
                                                    __bf16* __restrict__ out, int n) {
    int i = blockIdx.x * 256 + threadIdx.x;
    int stride = gridDim.x * 256;
    for (; i < n; i += stride) out[i] = (__bf16)in[i];
}

// ------------------- GEMM + bias + LN-stats (fused epilogue) ---------------
// C[m,n] = sum_k A[m,k]*W[n,k] + bias[n]   (A:[M,K] bf16, W:[N,K] bf16)
// stats[m] += {sum_n C, sum_n C^2}  via f32 atomics (for LayerNorm).
__global__ __launch_bounds__(256) void gemm_bias_stats(
    const __bf16* __restrict__ A, const __bf16* __restrict__ W,
    const float* __restrict__ bias, float* __restrict__ C,
    float* __restrict__ stats, int M, int N, int K)
{
    __shared__ __bf16 lA[2][TM * LSTR];
    __shared__ __bf16 lB[2][TN * LSTR];

    const int mblocks = M / TM;
    const int Mbase = (blockIdx.x % mblocks) * TM;
    const int Nbase = (blockIdx.x / mblocks) * TN;

    const int tid   = threadIdx.x;
    const int lane  = tid & 31;
    const int wave  = tid >> 5;
    const int waveM = wave & 3;   // 4 waves along M (32 rows each)
    const int waveN = wave >> 2;  // 2 waves along N (64 cols each)

    const int lr    = lane & 15;        // row/col within 16x16 tile
    const int khalf = (lane >> 4) * 8;  // K sub-chunk select per ISA layout

    // Per-thread staging chunks: 512 x 16B chunks per 128x32 tile pair,
    // 2 chunks of A + 2 chunks of W per thread per K-step.
    const int c0 = tid, c1 = tid + 256;
    const int r0 = c0 >> 2, ko0 = (c0 & 3) * 8;
    const int r1 = c1 >> 2, ko1 = (c1 & 3) * 8;
    const __bf16* gA0 = &A[(size_t)(Mbase + r0) * K + ko0];
    const __bf16* gA1 = &A[(size_t)(Mbase + r1) * K + ko1];
    const __bf16* gW0 = &W[(size_t)(Nbase + r0) * K + ko0];
    const __bf16* gW1 = &W[(size_t)(Nbase + r1) * K + ko1];
    const uint32_t lA0[2] = { lds_off32(&lA[0][r0 * LSTR + ko0]),
                              lds_off32(&lA[1][r0 * LSTR + ko0]) };
    const uint32_t lA1[2] = { lds_off32(&lA[0][r1 * LSTR + ko1]),
                              lds_off32(&lA[1][r1 * LSTR + ko1]) };
    const uint32_t lB0[2] = { lds_off32(&lB[0][r0 * LSTR + ko0]),
                              lds_off32(&lB[1][r0 * LSTR + ko0]) };
    const uint32_t lB1[2] = { lds_off32(&lB[0][r1 * LSTR + ko1]),
                              lds_off32(&lB[1][r1 * LSTR + ko1]) };

    v8f acc[2][4];
    #pragma unroll
    for (int i = 0; i < 2; ++i)
        #pragma unroll
        for (int j = 0; j < 4; ++j)
            acc[i][j] = (v8f)(0.0f);

    // prologue: DMA first K-step into buffer 0
    async_ld_b128(gA0, lA0[0]);
    async_ld_b128(gA1, lA1[0]);
    async_ld_b128(gW0, lB0[0]);
    async_ld_b128(gW1, lB1[0]);

    int cur = 0;
    for (int k0 = 0; k0 < K; k0 += TK) {
        const int nxt = cur ^ 1;
        if (k0 + TK < K) {
            // DMA next K-step into the alternate buffer (overlaps WMMA below)
            async_ld_b128(gA0 + k0 + TK, lA0[nxt]);
            async_ld_b128(gA1 + k0 + TK, lA1[nxt]);
            async_ld_b128(gW0 + k0 + TK, lB0[nxt]);
            async_ld_b128(gW1 + k0 + TK, lB1[nxt]);
            asm volatile("s_wait_asynccnt 0x4" ::: "memory");  // prev batch done
        } else {
            asm volatile("s_wait_asynccnt 0x0" ::: "memory");  // last batch done
        }
        __syncthreads();   // all waves' DMA writes to lA/lB[cur] visible

        // ---- fragments per CDNA5 16-bit A layout:
        //      lane<16: K in [0,8)+[16,24); lane>=16: K in [8,16)+[24,32) ----
        v16bf afr[2], bfr[4];
        #pragma unroll
        for (int i = 0; i < 2; ++i) {
            int row = waveM * 32 + i * 16 + lr;
            v8bf lo = *(const v8bf*)&lA[cur][row * LSTR + khalf];
            v8bf hi = *(const v8bf*)&lA[cur][row * LSTR + khalf + 16];
            #pragma unroll
            for (int e = 0; e < 8; ++e) { afr[i][e] = lo[e]; afr[i][e + 8] = hi[e]; }
        }
        #pragma unroll
        for (int j = 0; j < 4; ++j) {
            int col = waveN * 64 + j * 16 + lr;
            v8bf lo = *(const v8bf*)&lB[cur][col * LSTR + khalf];
            v8bf hi = *(const v8bf*)&lB[cur][col * LSTR + khalf + 16];
            #pragma unroll
            for (int e = 0; e < 8; ++e) { bfr[j][e] = lo[e]; bfr[j][e + 8] = hi[e]; }
        }

        #pragma unroll
        for (int i = 0; i < 2; ++i)
            #pragma unroll
            for (int j = 0; j < 4; ++j)
                acc[i][j] = __builtin_amdgcn_wmma_f32_16x16x32_bf16(
                    false, afr[i], false, bfr[j], (short)0, acc[i][j], false, false);

        __syncthreads();   // all waves done reading [cur] before it is re-targeted
        cur = nxt;
    }

    // ---- epilogue: bias add, NT store of fp32 intermediate, LN stats ----
    // C/D layout: lane<16 -> M=vg, N=lr ; lane>=16 -> M=8+vg, N=lr
    const int half8 = (lane >> 4) * 8;
    float rs[2][8], rss[2][8];
    #pragma unroll
    for (int i = 0; i < 2; ++i)
        #pragma unroll
        for (int vg = 0; vg < 8; ++vg) { rs[i][vg] = 0.0f; rss[i][vg] = 0.0f; }

    #pragma unroll
    for (int i = 0; i < 2; ++i) {
        int mg = Mbase + waveM * 32 + i * 16 + half8;
        #pragma unroll
        for (int j = 0; j < 4; ++j) {
            int n = Nbase + waveN * 64 + j * 16 + lr;
            float b = bias[n];
            #pragma unroll
            for (int vg = 0; vg < 8; ++vg) {
                float v = acc[i][j][vg] + b;
                __builtin_nontemporal_store(v, &C[(size_t)(mg + vg) * N + n]);
                rs[i][vg]  += v;
                rss[i][vg] += v * v;
            }
        }
    }

    // reduce over the 16 lanes of each half-group (distinct n, same m)
    #pragma unroll
    for (int off = 8; off >= 1; off >>= 1) {
        #pragma unroll
        for (int i = 0; i < 2; ++i)
            #pragma unroll
            for (int vg = 0; vg < 8; ++vg) {
                rs[i][vg]  += __shfl_xor(rs[i][vg],  off, 16);
                rss[i][vg] += __shfl_xor(rss[i][vg], off, 16);
            }
    }
    if (lr == 0) {
        #pragma unroll
        for (int i = 0; i < 2; ++i)
            #pragma unroll
            for (int vg = 0; vg < 8; ++vg) {
                int m = Mbase + waveM * 32 + i * 16 + half8 + vg;
                atomicAdd(&stats[2 * m + 0], rs[i][vg]);
                atomicAdd(&stats[2 * m + 1], rss[i][vg]);
            }
    }
}

// ------------------------- LayerNorm normalize -----------------------------
__global__ __launch_bounds__(256) void ln_normalize(
    const float* __restrict__ C, const float* __restrict__ stats,
    __bf16* __restrict__ actOut, float* __restrict__ f32Out, int N)
{
    int row = blockIdx.x;
    float inv  = 1.0f / (float)N;
    float mean = stats[2 * row + 0] * inv;
    float var  = stats[2 * row + 1] * inv - mean * mean;
    float rstd = rsqrtf(var + 1e-5f);
    const float* cr = C + (size_t)row * N;
    __bf16* ar = actOut + (size_t)row * N;
    if (f32Out) {
        float* fr = f32Out + (size_t)row * N;
        for (int i = threadIdx.x; i < N; i += 256) {
            float v = (__builtin_nontemporal_load(&cr[i]) - mean) * rstd;
            ar[i] = (__bf16)v;
            fr[i] = v;
        }
    } else {
        for (int i = threadIdx.x; i < N; i += 256) {
            float v = (__builtin_nontemporal_load(&cr[i]) - mean) * rstd;
            ar[i] = (__bf16)v;
        }
    }
}

// ---------------------------------------------------------------------------
extern "C" void kernel_launch(void* const* d_in, const int* in_sizes, int n_in,
                              void* d_out, int out_size, void* d_ws, size_t ws_size,
                              hipStream_t stream) {
    (void)in_sizes; (void)n_in; (void)out_size; (void)ws_size;
    const int M = 8 * 2048, DI = 1024, DO = 4096;

    const float* x        = (const float*)d_in[0];
    const float* w_in     = (const float*)d_in[1];   // [DO, DI]
    const float* b_in     = (const float*)d_in[2];   // [DO]
    const float* w_out    = (const float*)d_in[3];   // [DI, DO]
    const float* b_out    = (const float*)d_in[4];   // [DI]
    // d_in[5] = lengths: unused (dense baseline)

    char* ws = (char*)d_ws;
    size_t off = 0;
    __bf16* actA  = (__bf16*)(ws + off); off += (size_t)M * DI * 2;   // 32 MB
    __bf16* actB  = (__bf16*)(ws + off); off += (size_t)M * DO * 2;   // 128 MB
    __bf16* winB  = (__bf16*)(ws + off); off += (size_t)DO * DI * 2;  // 8 MB
    __bf16* woutB = (__bf16*)(ws + off); off += (size_t)DI * DO * 2;  // 8 MB
    float*  Cbuf  = (float*) (ws + off); off += (size_t)M * DO * 4;   // 256 MB
    float*  stats = (float*) (ws + off); off += (size_t)M * 2 * 4;    // 128 KB

    // one-time bf16 casts
    cvt_f32_bf16<<<1024, 256, 0, stream>>>(x,     actA,  M * DI);
    cvt_f32_bf16<<<1024, 256, 0, stream>>>(w_in,  winB,  DO * DI);
    cvt_f32_bf16<<<1024, 256, 0, stream>>>(w_out, woutB, DI * DO);

    const int g1 = (M / TM) * (DO / TN);  // 128 * 32
    const int g2 = (M / TM) * (DI / TN);  // 128 * 8

    for (int it = 0; it < 10; ++it) {
        // linear 1: [M,DI] x [DO,DI]^T -> [M,DO], + LN
        hipMemsetAsync(stats, 0, (size_t)M * 2 * 4, stream);
        gemm_bias_stats<<<g1, 256, 0, stream>>>(actA, winB, b_in, Cbuf, stats, M, DO, DI);
        ln_normalize<<<M, 256, 0, stream>>>(Cbuf, stats, actB, nullptr, DO);

        // linear 2: [M,DO] x [DI,DO]^T -> [M,DI], + LN
        hipMemsetAsync(stats, 0, (size_t)M * 2 * 4, stream);
        gemm_bias_stats<<<g2, 256, 0, stream>>>(actB, woutB, b_out, Cbuf, stats, M, DI, DO);
        ln_normalize<<<M, 256, 0, stream>>>(Cbuf, stats, actA,
                                            (it == 9) ? (float*)d_out : nullptr, DI);
    }
}
